// GroupedQueryAttention_77386720740073
// MI455X (gfx1250) — matrix operand order repaired and checked by reading, not compiled, not run on hardware
//
#include <hip/hip_runtime.h>
#include <hip/hip_bf16.h>

// ---------------------------------------------------------------------------
// Problem constants (from reference)
// ---------------------------------------------------------------------------
#define BATCH     2
#define SEQ       2048
#define HID       2048
#define NHEADS    16
#define NKV       4
#define HDIM      128
#define KVDIM     (NKV * HDIM)          // 512
#define MROWS     (BATCH * SEQ)         // 4096
#define SM_SCALE  0.08838834764831845f  // 1/sqrt(128)

typedef __attribute__((ext_vector_type(16))) __bf16 bfx16;
typedef __attribute__((ext_vector_type(8)))  float  floatx8;
typedef unsigned short u16;
typedef unsigned int   u32;

union Frag {            // one WMMA 16-bit A/B fragment: 16 bf16 per lane
    uint4 u[2];
    bfx16 v;
};

__device__ __forceinline__ floatx8 fzero8() {
    floatx8 z;
#pragma unroll
    for (int i = 0; i < 8; ++i) z[i] = 0.0f;
    return z;
}

__device__ __forceinline__ u16 f2bf(float f) {
    union { float f; u32 u; } x; x.f = f;
    u32 r = x.u + 0x7FFFu + ((x.u >> 16) & 1u);   // round-to-nearest-even
    return (u16)(r >> 16);
}

__device__ __forceinline__ floatx8 wmma_bf16(bfx16 a, bfx16 b, floatx8 c) {
    return __builtin_amdgcn_wmma_f32_16x16x32_bf16(
        /*neg_a=*/false, a, /*neg_b=*/false, b,
        /*c_mod=*/(short)0, c, /*reuse_a=*/false, /*reuse_b=*/false);
}

// ---------------------------------------------------------------------------
// fp32 -> bf16 cast (vectorized, 4 elems/thread)
// ---------------------------------------------------------------------------
__global__ __launch_bounds__(256) void cast_f32_bf16_kernel(const float* __restrict__ src,
                                                            u16* __restrict__ dst, int n4) {
    int i = blockIdx.x * blockDim.x + threadIdx.x;
    if (i >= n4) return;
    float4 f = reinterpret_cast<const float4*>(src)[i];
    ushort4 o;
    o.x = f2bf(f.x); o.y = f2bf(f.y); o.z = f2bf(f.z); o.w = f2bf(f.w);
    reinterpret_cast<ushort4*>(dst)[i] = o;
}

// ---------------------------------------------------------------------------
// Tiled bf16 WMMA GEMM:  Y[M,N] = X[M,K] @ W[N,K]^T + bias, optional fused RoPE
// Block tile 128x128, BK=64 K-slabs, LDS ping-pong double-buffer: next slab is
// copied global->LDS(buf^1) with short-lived regs while the current slab's 16
// WMMAs read LDS(buf); exactly one barrier per slab. 8 waves of 2x4 tiles.
// ---------------------------------------------------------------------------
__global__ __launch_bounds__(256) void gemm_bf16_wmma_kernel(
        const u16* __restrict__ X, const u16* __restrict__ W,
        const float* __restrict__ bias,
        u16* __restrict__ outb, float* __restrict__ outf,
        int M, int N, int K, int S, int rope)
{
    __shared__ __align__(16) u16 Xs[2][128 * 64];   // 2 x 16 KB
    __shared__ __align__(16) u16 Ws[2][128 * 64];   // 2 x 16 KB

    const int m0   = blockIdx.x * 128;
    const int n0   = blockIdx.y * 128;
    const int tid  = threadIdx.x;
    const int lane = tid & 31;
    const int w    = tid >> 5;
    const int wm   = w & 3;          // 4 waves along M, 32 rows each
    const int wn   = w >> 2;         // 2 waves along N, 64 cols each

    floatx8 acc[2][4];
#pragma unroll
    for (int mt = 0; mt < 2; ++mt)
#pragma unroll
        for (int nt = 0; nt < 4; ++nt) acc[mt][nt] = fzero8();

    const int rsel = lane & 15;
    const int koA  = (lane >> 4) * 8;   // A-frag K offset (ISA 16-bit A layout)
    const int koB  = (lane >> 4) * 8;   // B-frag K offset (ISA 16-bit B layout)

    // staging geometry: thread covers rows rbase + {0,32,64,96}, one 8-elem chunk
    const int rbase = tid >> 3;          // 0..31
    const int ch    = (tid & 7) * 8;     // 0..56
    const size_t gstride = (size_t)32 * K;       // row stride between chunks
    const u16* gx0 = X + (size_t)(m0 + rbase) * K + ch;
    const u16* gw0 = W + (size_t)(n0 + rbase) * K + ch;
    const int lbase = rbase * 64 + ch;

    // copy one 128x64 slab pair global -> LDS[buf]; transient registers only
    auto stage = [&](int kt, int buf) {
#pragma unroll
        for (int i = 0; i < 4; ++i) {
            const uint4 vx = *reinterpret_cast<const uint4*>(gx0 + kt + (size_t)i * gstride);
            const uint4 vw = *reinterpret_cast<const uint4*>(gw0 + kt + (size_t)i * gstride);
            *reinterpret_cast<uint4*>(&Xs[buf][lbase + i * 32 * 64]) = vx;
            *reinterpret_cast<uint4*>(&Ws[buf][lbase + i * 32 * 64]) = vw;
        }
    };
    // 16 WMMAs on LDS[buf]
    auto compute = [&](int buf) {
        const u16* xs = &Xs[buf][0];
        const u16* wsv = &Ws[buf][0];
#pragma unroll
        for (int ks = 0; ks < 64; ks += 32) {
            Frag a[2], bf[4];
#pragma unroll
            for (int mt = 0; mt < 2; ++mt) {
                const int row = wm * 32 + mt * 16 + rsel;
                a[mt].u[0] = *reinterpret_cast<const uint4*>(&xs[row * 64 + ks + koA]);
                a[mt].u[1] = *reinterpret_cast<const uint4*>(&xs[row * 64 + ks + koA + 16]);
            }
#pragma unroll
            for (int nt = 0; nt < 4; ++nt) {
                const int row = wn * 64 + nt * 16 + rsel;
                bf[nt].u[0] = *reinterpret_cast<const uint4*>(&wsv[row * 64 + ks + koB]);
                bf[nt].u[1] = *reinterpret_cast<const uint4*>(&wsv[row * 64 + ks + koB + 16]);
            }
#pragma unroll
            for (int mt = 0; mt < 2; ++mt)
#pragma unroll
                for (int nt = 0; nt < 4; ++nt)
                    acc[mt][nt] = wmma_bf16(a[mt].v, bf[nt].v, acc[mt][nt]);
        }
    };

    stage(0, 0);
    __syncthreads();

    int buf = 0;
    for (int kt = 0; kt < K - 64; kt += 64) {
        stage(kt + 64, buf ^ 1);              // feeds NEXT iteration
        __builtin_prefetch(gx0 + kt + 128, 0, 1);   // gfx1250 global_prefetch
        __builtin_prefetch(gw0 + kt + 128, 0, 1);
        compute(buf);                         // hides the copy latency
        __syncthreads();                      // one barrier per slab
        buf ^= 1;
    }
    compute(buf);                             // peeled last slab (no copy)

    // epilogue: bias + optional fused interleaved RoPE + store
    const int hi = lane >> 4;
#pragma unroll
    for (int mt = 0; mt < 2; ++mt) {
#pragma unroll
        for (int nt = 0; nt < 4; ++nt) {
            const int colb = n0 + wn * 64 + nt * 16 + rsel;
            const float bia = bias[colb];
#pragma unroll
            for (int r = 0; r < 8; ++r) {
                const int row = m0 + wm * 32 + mt * 16 + r + hi * 8;
                float v = acc[mt][nt][r] + bia;
                if (rope) {
                    const int pos = row % S;                 // position within sequence
                    const int i2  = (colb & (HDIM - 1)) >> 1;
                    const float invf = __powf(10000.0f, -(float)(2 * i2) * (1.0f / (float)HDIM));
                    const float ang  = (float)pos * invf;
                    float sn, cs;
                    __sincosf(ang, &sn, &cs);
                    // pair (2i,2i+1) sits on adjacent lanes in the C layout
                    const float partner = __shfl_xor(v, 1, 32);
                    v = v * cs + ((colb & 1) ? partner : -partner) * sn;
                }
                if (outf) outf[(size_t)row * N + colb] = v;
                else      outb[(size_t)row * N + colb] = f2bf(v);
            }
        }
    }
}

// ---------------------------------------------------------------------------
// V transpose: v[(b*S+s)*KVDIM + kvh*HDIM + d]  ->  vt[((b*NKV+kvh)*HDIM+d)*S + s]
// ---------------------------------------------------------------------------
__global__ __launch_bounds__(256) void transpose_v_kernel(const u16* __restrict__ v,
                                                          u16* __restrict__ vt, int total) {
    int i = blockIdx.x * blockDim.x + threadIdx.x;
    if (i >= total) return;
    const int s    = i & (SEQ - 1);
    const int d    = (i >> 11) & (HDIM - 1);
    const int kvh  = (i >> 18) & (NKV - 1);
    const int b    = i >> 20;
    vt[i] = v[((size_t)b * SEQ + s) * KVDIM + kvh * HDIM + d];
}

// ---------------------------------------------------------------------------
// Causal GQA flash attention: 4 waves/block, each wave owns one 16-row Q tile.
// QK^T and PV both on v_wmma_f32_16x16x32_bf16; online softmax in fp32.
// ---------------------------------------------------------------------------
__global__ __launch_bounds__(128) void flash_attn_wmma_kernel(
        const u16* __restrict__ qb, const u16* __restrict__ kb,
        const u16* __restrict__ vt, u16* __restrict__ att)
{
    __shared__ __align__(16) u16 Pl[4 * 16 * 32];   // per-wave P staging (C->A relayout)

    const int b    = blockIdx.z;
    const int h    = blockIdx.y;
    const int kvh  = h / (NHEADS / NKV);
    const int wave = threadIdx.x >> 5;
    const int lane = threadIdx.x & 31;
    const int hi   = lane >> 4;
    const int rsel = lane & 15;
    const int koA  = hi * 8;
    const int koB  = hi * 8;
    const int qs   = (blockIdx.x * 4 + wave) * 16;

    u16* pw = &Pl[wave * 512];

    // Q fragments for this wave's 16 rows (head_dim=128 -> 4 K-chunks of 32)
    Frag aq[4];
    {
        const size_t qrow = ((size_t)b * SEQ + qs + rsel) * HID + (size_t)h * HDIM;
#pragma unroll
        for (int kc = 0; kc < 4; ++kc) {
            aq[kc].u[0] = *reinterpret_cast<const uint4*>(&qb[qrow + kc * 32 + koA]);
            aq[kc].u[1] = *reinterpret_cast<const uint4*>(&qb[qrow + kc * 32 + koA + 16]);
        }
    }

    floatx8 o[8];
#pragma unroll
    for (int dt = 0; dt < 8; ++dt) o[dt] = fzero8();
    float mrow[8], lrow[8];
#pragma unroll
    for (int r = 0; r < 8; ++r) { mrow[r] = -__builtin_inff(); lrow[r] = 0.0f; }

    const int kend = qs + 16;   // causal: keys 0..qs+15
    for (int kb0 = 0; kb0 < kend; kb0 += 32) {
        // ---- scores: S(16x32) = Q(16x128) @ K^T, two 16-wide N tiles ----
        floatx8 sc[2];
#pragma unroll
        for (int nt = 0; nt < 2; ++nt) {
            const int key = kb0 + nt * 16 + rsel;
            const size_t krow = ((size_t)b * SEQ + key) * KVDIM + (size_t)kvh * HDIM;
            floatx8 s = fzero8();
#pragma unroll
            for (int kc = 0; kc < 4; ++kc) {
                Frag bk;
                bk.u[0] = *reinterpret_cast<const uint4*>(&kb[krow + kc * 32 + koB]);
                bk.u[1] = *reinterpret_cast<const uint4*>(&kb[krow + kc * 32 + koB + 16]);
                s = wmma_bf16(aq[kc].v, bk.v, s);
            }
            sc[nt] = s * SM_SCALE;
        }
        // ---- causal mask ----
#pragma unroll
        for (int nt = 0; nt < 2; ++nt) {
            const int col = kb0 + nt * 16 + rsel;
#pragma unroll
            for (int r = 0; r < 8; ++r) {
                const int row = qs + r + hi * 8;
                if (col > row) sc[nt][r] = -__builtin_inff();
            }
        }
        // ---- online softmax (row reductions across 16 lanes) ----
#pragma unroll
        for (int r = 0; r < 8; ++r) {
            float v0 = sc[0][r], v1 = sc[1][r];
            float mx = fmaxf(v0, v1);
#pragma unroll
            for (int msk = 1; msk < 16; msk <<= 1)
                mx = fmaxf(mx, __shfl_xor(mx, msk, 16));
            const float mn    = fmaxf(mrow[r], mx);
            const float alpha = __expf(mrow[r] - mn);
            const float p0 = __expf(v0 - mn);
            const float p1 = __expf(v1 - mn);
            float rs = p0 + p1;
#pragma unroll
            for (int msk = 1; msk < 16; msk <<= 1)
                rs += __shfl_xor(rs, msk, 16);
            lrow[r] = lrow[r] * alpha + rs;
            mrow[r] = mn;
#pragma unroll
            for (int dt = 0; dt < 8; ++dt) o[dt][r] *= alpha;
            pw[(r + hi * 8) * 32 + rsel]      = f2bf(p0);
            pw[(r + hi * 8) * 32 + 16 + rsel] = f2bf(p1);
        }
        // LDS ops are in-order per wave; drain DScnt before re-reading P
        asm volatile("s_wait_dscnt 0x0" ::: "memory");

        // ---- P in A layout, then O(16x128) += P(16x32) @ V(32x128) ----
        Frag ap;
        ap.u[0] = *reinterpret_cast<const uint4*>(&pw[rsel * 32 + koA]);
        ap.u[1] = *reinterpret_cast<const uint4*>(&pw[rsel * 32 + koA + 16]);
#pragma unroll
        for (int dt = 0; dt < 8; ++dt) {
            const int dim = dt * 16 + rsel;
            const size_t vrow = (((size_t)b * NKV + kvh) * HDIM + dim) * SEQ + kb0 + koB;
            Frag bv;
            bv.u[0] = *reinterpret_cast<const uint4*>(&vt[vrow]);
            bv.u[1] = *reinterpret_cast<const uint4*>(&vt[vrow + 16]);
            o[dt] = wmma_bf16(ap.v, bv.v, o[dt]);
        }
    }

    // ---- normalize and store attended (bf16) ----
#pragma unroll
    for (int dt = 0; dt < 8; ++dt) {
#pragma unroll
        for (int r = 0; r < 8; ++r) {
            const int row = qs + r + hi * 8;
            const float val = o[dt][r] / lrow[r];
            att[((size_t)b * SEQ + row) * HID + (size_t)h * HDIM + dt * 16 + rsel] = f2bf(val);
        }
    }
}

// ---------------------------------------------------------------------------
// Host-side orchestration
// ---------------------------------------------------------------------------
extern "C" void kernel_launch(void* const* d_in, const int* in_sizes, int n_in,
                              void* d_out, int out_size, void* d_ws, size_t ws_size,
                              hipStream_t stream) {
    (void)in_sizes; (void)n_in; (void)out_size; (void)ws_size;

    const float* x  = (const float*)d_in[0];
    const float* Wq = (const float*)d_in[1];
    const float* bq = (const float*)d_in[2];
    const float* Wk = (const float*)d_in[3];
    const float* bk = (const float*)d_in[4];
    const float* Wv = (const float*)d_in[5];
    const float* bv = (const float*)d_in[6];
    const float* Wo = (const float*)d_in[7];
    const float* bo = (const float*)d_in[8];
    float* out = (float*)d_out;

    // workspace layout (bf16 buffers)
    char* ws = (char*)d_ws;
    size_t off = 0;
    u16* xb  = (u16*)(ws + off); off += (size_t)MROWS * HID   * 2;  // 16 MB
    u16* wqb = (u16*)(ws + off); off += (size_t)HID   * HID   * 2;  //  8 MB
    u16* wkb = (u16*)(ws + off); off += (size_t)KVDIM * HID   * 2;  //  2 MB
    u16* wvb = (u16*)(ws + off); off += (size_t)KVDIM * HID   * 2;  //  2 MB
    u16* wob = (u16*)(ws + off); off += (size_t)HID   * HID   * 2;  //  8 MB
    u16* qb  = (u16*)(ws + off); off += (size_t)MROWS * HID   * 2;  // 16 MB
    u16* kbf = (u16*)(ws + off); off += (size_t)MROWS * KVDIM * 2;  //  4 MB
    u16* vbf = (u16*)(ws + off); off += (size_t)MROWS * KVDIM * 2;  //  4 MB
    u16* vtb = (u16*)(ws + off); off += (size_t)MROWS * KVDIM * 2;  //  4 MB
    u16* att = (u16*)(ws + off); off += (size_t)MROWS * HID   * 2;  // 16 MB

    // 1) casts to bf16
    {
        const int nx = MROWS * HID / 4;
        cast_f32_bf16_kernel<<<(nx + 255) / 256, 256, 0, stream>>>(x, xb, nx);
        const int nq = HID * HID / 4;
        cast_f32_bf16_kernel<<<(nq + 255) / 256, 256, 0, stream>>>(Wq, wqb, nq);
        const int nk = KVDIM * HID / 4;
        cast_f32_bf16_kernel<<<(nk + 255) / 256, 256, 0, stream>>>(Wk, wkb, nk);
        cast_f32_bf16_kernel<<<(nk + 255) / 256, 256, 0, stream>>>(Wv, wvb, nk);
        cast_f32_bf16_kernel<<<(nq + 255) / 256, 256, 0, stream>>>(Wo, wob, nq);
    }

    // 2) Q/K/V projections (RoPE fused into Q and K epilogues)
    gemm_bf16_wmma_kernel<<<dim3(MROWS / 128, HID / 128),   256, 0, stream>>>(
        xb, wqb, bq, qb,  nullptr, MROWS, HID,   HID, SEQ, /*rope=*/1);
    gemm_bf16_wmma_kernel<<<dim3(MROWS / 128, KVDIM / 128), 256, 0, stream>>>(
        xb, wkb, bk, kbf, nullptr, MROWS, KVDIM, HID, SEQ, /*rope=*/1);
    gemm_bf16_wmma_kernel<<<dim3(MROWS / 128, KVDIM / 128), 256, 0, stream>>>(
        xb, wvb, bv, vbf, nullptr, MROWS, KVDIM, HID, SEQ, /*rope=*/0);

    // 3) V transpose for contiguous PV B-fragments
    {
        const int nt = MROWS * KVDIM;
        transpose_v_kernel<<<(nt + 255) / 256, 256, 0, stream>>>(vbf, vtb, nt);
    }

    // 4) causal GQA flash attention
    flash_attn_wmma_kernel<<<dim3(SEQ / 64, NHEADS, BATCH), 128, 0, stream>>>(
        qb, kbf, vtb, att);

    // 5) output projection (fp32 out + bias)
    gemm_bf16_wmma_kernel<<<dim3(MROWS / 128, HID / 128), 256, 0, stream>>>(
        att, wob, bo, nullptr, out, MROWS, HID, HID, SEQ, /*rope=*/0);
}
